// TrajectoryDecoder_2018634629625
// MI455X (gfx1250) — matrix operand order repaired and checked by reading, not compiled, not run on hardware
//
#include <hip/hip_runtime.h>

// ---------------------------------------------------------------------------
// Fused teacher-forced LSTM decoder for MI455X (gfx1250, wave32, WMMA).
//   N=32768 rows, T=12 steps, P=2, H=256, E=128, 4H=1024
// Block = 512 threads (16 waves) owning 64 batch rows for ALL 12 steps.
//   - wave w owns hh in [16w,16w+16); every B fragment reused by 4 WMMAs
//   - gates split into two passes (i,g then f,o): only 8 C-tiles live at once
//   - K-chunk loops ROLLED with loop-carried pointers so the compiler cannot
//     hoist/prefetch all B fragments at once (this caused scratch spills)
//   - amdgpu_waves_per_eu(4,4) pins the 256-VGPR budget for 16 waves/WGP
// ---------------------------------------------------------------------------

typedef __bf16 bf16;
typedef __attribute__((ext_vector_type(16))) __bf16 v16bf;
typedef __attribute__((ext_vector_type(8)))  __bf16 v8bf;
typedef __attribute__((ext_vector_type(8)))  float  v8f;

#define N_TOT    32768
#define T_STEPS  12
#define P_DIM    2
#define H_DIM    256
#define E_DIM    128
#define FOURH    1024

#define M_BLK    64          // batch rows per block
#define NTHREADS 512         // 16 waves

// per-wave weight regions (elements): fragment(kcidx,g) = 32 lanes * 16 bf16
#define FRAG_ELEMS (32 * 16)                   // 512
#define RK_WSTRIDE (8 * 4 * FRAG_ELEMS)        // 16384 (8 k-chunks x 4 gates)
#define KK_WSTRIDE (4 * 4 * FRAG_ELEMS)        // 8192  (4 k-chunks x 4 gates)
#define RK_ELEMS   (16 * RK_WSTRIDE)           // 262144
#define KK_ELEMS   (16 * KK_WSTRIDE)           // 131072

__device__ __forceinline__ bf16 f2bf(float f) {
    union { float f; unsigned u; } un; un.f = f;
    unsigned r = un.u + 0x7fffu + ((un.u >> 16) & 1u);   // round-nearest-even
    return __builtin_bit_cast(bf16, (unsigned short)(r >> 16));
}
__device__ __forceinline__ float bf2f(bf16 b) {
    unsigned u = ((unsigned)__builtin_bit_cast(unsigned short, b)) << 16;
    return __builtin_bit_cast(float, u);
}
__device__ __forceinline__ float sigmoidf(float x) {
    return 1.0f / (1.0f + __expf(-x));
}

// f32 weights -> bf16 per-wave fragment layout:
//  rkernel[k][n]: g=n>>8, hh=n&255, w=hh>>4, nn=hh&15, kcidx=k>>5, kl=k&31
//    -> ws[w*RK_WSTRIDE + (kcidx*4+g)*512 + (((kl>>3)&1)*16+nn)*16 + (kl&7) + ((kl>>4)<<3)]      (kernel likewise)
__global__ __launch_bounds__(256) void convert_weights_kernel(
    const float* __restrict__ kern, const float* __restrict__ rkern,
    bf16* __restrict__ ws)
{
    int idx = blockIdx.x * 256 + threadIdx.x;
    if (idx < H_DIM * FOURH) {
        int k = idx >> 10, n = idx & 1023;
        int g = n >> 8, hh = n & 255, w = hh >> 4, nn = hh & 15;
        int kcidx = k >> 5, kl = k & 31;
        ws[(size_t)w * RK_WSTRIDE + (kcidx * 4 + g) * FRAG_ELEMS + (((kl >> 3) & 1) * 16 + nn) * 16 + (kl & 7) + ((kl >> 4) << 3)]
            = f2bf(rkern[idx]);
    }
    if (idx < E_DIM * FOURH) {
        int k = idx >> 10, n = idx & 1023;
        int g = n >> 8, hh = n & 255, w = hh >> 4, nn = hh & 15;
        int kcidx = k >> 5, kl = k & 31;
        ws[RK_ELEMS + (size_t)w * KK_WSTRIDE + (kcidx * 4 + g) * FRAG_ELEMS + (((kl >> 3) & 1) * 16 + nn) * 16 + (kl & 7) + ((kl >> 4) << 3)]
            = f2bf(kern[idx]);
    }
}

// One pass over the full K dimension accumulating TWO gates x 4 M-tiles.
// acc[0..3] = first gate (mi 0..3), acc[4..7] = second gate.
// ROLLED loops + loop-carried pointers: bounds prefetch depth, kills spills.
__device__ __forceinline__ void gemm_pass(
    v8f acc[8],
    const bf16* __restrict__ h_base, const bf16* __restrict__ e_base,
    const bf16* __restrict__ brg0, const bf16* __restrict__ brg1,
    const bf16* __restrict__ bkg0, const bf16* __restrict__ bkg1)
{
    const bf16* hp = h_base;
    const bf16* p0 = brg0;
    const bf16* p1 = brg1;
    #pragma unroll 1
    for (int kcidx = 0; kcidx < 8; ++kcidx) {        // recurrent, K = 256
        v16bf a[4];
        #pragma unroll
        for (int mi = 0; mi < 4; ++mi) {
            const bf16* p = hp + mi * 16 * H_DIM;
            v8bf lo = *(const v8bf*)p;
            v8bf hi = *(const v8bf*)(p + 16);
            a[mi] = __builtin_shufflevector(lo, hi,
                        0,1,2,3,4,5,6,7,8,9,10,11,12,13,14,15);
        }
        v16bf b0 = *(const v16bf*)p0;
        v16bf b1 = *(const v16bf*)p1;
        #pragma unroll
        for (int mi = 0; mi < 4; ++mi)
            acc[mi] = __builtin_amdgcn_wmma_f32_16x16x32_bf16(
                false, a[mi], false, b0, (short)0, acc[mi], false, false);
        #pragma unroll
        for (int mi = 0; mi < 4; ++mi)
            acc[4 + mi] = __builtin_amdgcn_wmma_f32_16x16x32_bf16(
                false, a[mi], false, b1, (short)0, acc[4 + mi], false, false);
        hp += 32;
        p0 += 4 * FRAG_ELEMS;
        p1 += 4 * FRAG_ELEMS;
    }
    const bf16* ep = e_base;
    p0 = bkg0;
    p1 = bkg1;
    #pragma unroll 1
    for (int kcidx = 0; kcidx < 4; ++kcidx) {        // input side, K = 128
        v16bf a[4];
        #pragma unroll
        for (int mi = 0; mi < 4; ++mi) {
            const bf16* p = ep + mi * 16 * E_DIM;
            v8bf lo = *(const v8bf*)p;
            v8bf hi = *(const v8bf*)(p + 16);
            a[mi] = __builtin_shufflevector(lo, hi,
                        0,1,2,3,4,5,6,7,8,9,10,11,12,13,14,15);
        }
        v16bf b0 = *(const v16bf*)p0;
        v16bf b1 = *(const v16bf*)p1;
        #pragma unroll
        for (int mi = 0; mi < 4; ++mi)
            acc[mi] = __builtin_amdgcn_wmma_f32_16x16x32_bf16(
                false, a[mi], false, b0, (short)0, acc[mi], false, false);
        #pragma unroll
        for (int mi = 0; mi < 4; ++mi)
            acc[4 + mi] = __builtin_amdgcn_wmma_f32_16x16x32_bf16(
                false, a[mi], false, b1, (short)0, acc[4 + mi], false, false);
        ep += 32;
        p0 += 4 * FRAG_ELEMS;
        p1 += 4 * FRAG_ELEMS;
    }
}

__global__ __launch_bounds__(NTHREADS)
__attribute__((amdgpu_waves_per_eu(4, 4)))
void lstm_decoder_kernel(
    const float* __restrict__ first_input,   // [N,2]
    const float* __restrict__ enc_h0,        // [N,H]
    const float* __restrict__ enc_c0,        // [N,H]
    const float* __restrict__ traj_gt,       // [N,T,2]
    const float* __restrict__ W_emb,         // [2,E]
    const float* __restrict__ b_emb,         // [E]
    const float* __restrict__ bias,          // [4H]
    const float* __restrict__ W_xy,          // [H,2]
    const float* __restrict__ b_xy,          // [2]
    const bf16*  __restrict__ wsw,           // converted weights
    float* __restrict__ out)                 // [N,T,2]
{
    __shared__ __align__(32) bf16  h_lds[M_BLK][H_DIM];     // 32 KB
    __shared__ __align__(32) bf16  emb_lds[M_BLK][E_DIM];   // 16 KB
    __shared__ __align__(32) float wxy_lds[P_DIM][H_DIM];   // 2 KB
    __shared__ float y_lds[M_BLK][P_DIM];                   // 512 B

    const int tid  = threadIdx.x;
    const int wave = tid >> 5;          // 0..15 : owns hh in [16w, 16w+16)
    const int lane = tid & 31;
    const int lrow = lane & 15;
    const int lhi  = lane >> 4;
    const int m0   = blockIdx.x * M_BLK;
    const int hh   = wave * 16 + lrow;  // this lane's single hh column

    const bf16* br = wsw + (size_t)wave * RK_WSTRIDE + lane * 16;
    const bf16* bk = wsw + RK_ELEMS + (size_t)wave * KK_WSTRIDE + lane * 16;
    const bf16* br_i = br + 0 * FRAG_ELEMS;   // gate order i,f,g,o
    const bf16* br_f = br + 1 * FRAG_ELEMS;
    const bf16* br_g = br + 2 * FRAG_ELEMS;
    const bf16* br_o = br + 3 * FRAG_ELEMS;
    const bf16* bk_i = bk + 0 * FRAG_ELEMS;
    const bf16* bk_f = bk + 1 * FRAG_ELEMS;
    const bf16* bk_g = bk + 2 * FRAG_ELEMS;
    const bf16* bk_o = bk + 3 * FRAG_ELEMS;

    const bf16* h_base = &h_lds[lrow][lhi * 8];
    const bf16* e_base = &emb_lds[lrow][lhi * 8];

    // persistent cell state + per-lane constants
    float creg[4][8];                   // [mi][v], row = mi*16 + v + 8*lhi
    float biasr[4];
    #pragma unroll
    for (int g = 0; g < 4; ++g) biasr[g] = bias[g * H_DIM + hh];
    #pragma unroll
    for (int mi = 0; mi < 4; ++mi)
        #pragma unroll
        for (int v = 0; v < 8; ++v)
            creg[mi][v] = enc_c0[(size_t)(m0 + mi * 16 + v + 8 * lhi) * H_DIM + hh];

    #pragma unroll
    for (int i = 0; i < (M_BLK * H_DIM) / NTHREADS; ++i) {
        int idx = i * NTHREADS + tid;
        int r = idx >> 8, c = idx & 255;
        h_lds[r][c] = f2bf(enc_h0[(size_t)(m0 + r) * H_DIM + c]);
    }
    { int p = tid >> 8, c = tid & 255; wxy_lds[p][c] = W_xy[c * 2 + p]; }
    if (tid < M_BLK * P_DIM) y_lds[tid >> 1][tid & 1] = 0.0f;

    const float by0 = b_xy[0], by1 = b_xy[1];

    #pragma unroll 1
    for (int t = 0; t < T_STEPS; ++t) {
        // ---- phase A: emb = relu(x @ W_emb + b_emb) ---------------------
        #pragma unroll
        for (int i = 0; i < (M_BLK * E_DIM) / NTHREADS; ++i) {
            int idx = i * NTHREADS + tid;
            int r = idx >> 7, e = idx & 127;
            size_t nrow = (size_t)(m0 + r);
            float x0, x1;
            if (t == 0) {
                x0 = first_input[nrow * 2 + 0];
                x1 = first_input[nrow * 2 + 1];
            } else {
                x0 = traj_gt[(nrow * T_STEPS + t) * 2 + 0];
                x1 = traj_gt[(nrow * T_STEPS + t) * 2 + 1];
            }
            float v = fmaf(x0, W_emb[e], fmaf(x1, W_emb[E_DIM + e], b_emb[e]));
            emb_lds[r][e] = f2bf(fmaxf(v, 0.0f));
        }
        __syncthreads();                       // emb + prev h ready

        // ---- pass 1: gates i and g -> tmp = sigma(zi)*tanh(zg) ----------
        float tmp[4][8];
        {
            v8f acc[8] = {};
            gemm_pass(acc, h_base, e_base, br_i, br_g, bk_i, bk_g);
            #pragma unroll
            for (int mi = 0; mi < 4; ++mi)
                #pragma unroll
                for (int v = 0; v < 8; ++v) {
                    float zi = acc[mi][v]     + biasr[0];
                    float zg = acc[4 + mi][v] + biasr[2];
                    tmp[mi][v] = sigmoidf(zi) * tanhf(zg);
                }
        }
        // ---- pass 2: gates f and o -> c, h ------------------------------
        float hval[4][8];
        {
            v8f acc[8] = {};
            gemm_pass(acc, h_base, e_base, br_f, br_o, bk_f, bk_o);
            #pragma unroll
            for (int mi = 0; mi < 4; ++mi)
                #pragma unroll
                for (int v = 0; v < 8; ++v) {
                    float zf = acc[mi][v]     + biasr[1];
                    float zo = acc[4 + mi][v] + biasr[3];
                    float c  = sigmoidf(zf) * creg[mi][v] + tmp[mi][v];
                    creg[mi][v] = c;
                    hval[mi][v] = sigmoidf(zo) * tanhf(c);
                }
        }
        __syncthreads();                       // all h_lds reads complete

        // ---- phase C: publish new h -------------------------------------
        #pragma unroll
        for (int mi = 0; mi < 4; ++mi)
            #pragma unroll
            for (int v = 0; v < 8; ++v)
                h_lds[mi * 16 + v + 8 * lhi][hh] = f2bf(hval[mi][v]);
        __syncthreads();                       // h writes visible

        // ---- phase D: y = h @ W_xy (cooperative, vectorized) ------------
        {
            int drow = tid >> 3;               // 0..63
            int dp   = (tid >> 2) & 1;
            int dseg = tid & 3;                // 64-hh segment
            float part = 0.0f;
            #pragma unroll
            for (int c8 = 0; c8 < 8; ++c8) {
                int h0 = dseg * 64 + c8 * 8;
                v8bf hv = *(const v8bf*)&h_lds[drow][h0];
                v8f  wv = *(const v8f*)&wxy_lds[dp][h0];
                #pragma unroll
                for (int e = 0; e < 8; ++e)
                    part = fmaf(bf2f(hv[e]), wv[e], part);
            }
            atomicAdd(&y_lds[drow][dp], part);
        }
        __syncthreads();                       // atomics done

        // ---- phase E: emit + re-arm ------------------------------------
        if (tid < M_BLK * P_DIM) {
            int r = tid >> 1, p = tid & 1;
            out[((size_t)(m0 + r) * T_STEPS + t) * P_DIM + p]
                = y_lds[r][p] + ((p == 0) ? by0 : by1);
            y_lds[r][p] = 0.0f;                // next atomics are 3 barriers away
        }
    }
}

extern "C" void kernel_launch(void* const* d_in, const int* in_sizes, int n_in,
                              void* d_out, int out_size, void* d_ws, size_t ws_size,
                              hipStream_t stream) {
    const float* first_input = (const float*)d_in[0];
    const float* enc_h0      = (const float*)d_in[1];
    const float* enc_c0      = (const float*)d_in[2];
    // d_in[3] = enc_h (unused by reference)
    const float* traj_gt     = (const float*)d_in[4];
    const float* W_emb       = (const float*)d_in[5];
    const float* b_emb       = (const float*)d_in[6];
    const float* kern        = (const float*)d_in[7];
    const float* rkern       = (const float*)d_in[8];
    const float* bias        = (const float*)d_in[9];
    const float* W_xy        = (const float*)d_in[10];
    const float* b_xy        = (const float*)d_in[11];
    bf16* wsw = (bf16*)d_ws;   // needs (RK_ELEMS + KK_ELEMS) * 2 = 768 KB

    convert_weights_kernel<<<(H_DIM * FOURH + 255) / 256, 256, 0, stream>>>(
        kern, rkern, wsw);

    lstm_decoder_kernel<<<N_TOT / M_BLK, NTHREADS, 0, stream>>>(
        first_input, enc_h0, enc_c0, traj_gt, W_emb, b_emb, bias, W_xy, b_xy,
        wsw, (float*)d_out);
}
